// DecoderWithAttention_23622320128252
// MI455X (gfx1250) — matrix operand order repaired and checked by reading, not compiled, not run on hardware
//
#include <hip/hip_runtime.h>
#include <hip/hip_bf16.h>
#include <stdint.h>

// Problem constants
constexpr int kB  = 128;
constexpr int kP  = 49;
constexpr int kH  = 512;
constexpr int kE  = 512;
constexpr int kV  = 10000;
constexpr int kA  = 49;
constexpr int kIF = 2048;
constexpr int kL  = 21;
constexpr int kT  = 20;           // decode steps = L-1
constexpr int kG  = 5 * kH;       // 2560 gate outputs
constexpr int kKxh = 2 * kE + kH; // 1536 = [emb, global, h]

typedef __attribute__((ext_vector_type(16))) _Float16 v16h;
typedef __attribute__((ext_vector_type(8)))  _Float16 v8h;
typedef __attribute__((ext_vector_type(8)))  float    v8f;
typedef int v4i_vs __attribute__((vector_size(16)));   // matches builtin param

// --------------------------------------------------------------------------
// Async global->LDS staging (CDNA5). Guarded so compilation never breaks.
// --------------------------------------------------------------------------
#if defined(__has_builtin)
#  if __has_builtin(__builtin_amdgcn_global_load_async_to_lds_b128)
#    define HAVE_ASYNC_LDS 1
#  endif
#endif
#ifndef HAVE_ASYNC_LDS
#  define HAVE_ASYNC_LDS 0
#endif

__device__ __forceinline__ void stage_b128(const _Float16* g, _Float16* l)
{
#if HAVE_ASYNC_LDS
    __builtin_amdgcn_global_load_async_to_lds_b128((v4i_vs*)g, (v4i_vs*)l, 0, 0);
#else
    *(v8h*)l = *(const v8h*)g;   // global_load_b128 + ds_store_b128
#endif
}

__device__ __forceinline__ void stage_wait()
{
#if HAVE_ASYNC_LDS
    asm volatile("s_wait_asynccnt 0x0" ::: "memory");
#endif
}

// ---------------------------------------------------------------------------
// WMMA GEMM: Out[M,N] = In[M,K] (f16, row-major, ld=ldin) @ W[N,K]^T (f16,
// row-major, ld=ldw) + bias[N]; optional per-row mask (t < dl[row]).
// Block = 256 threads = 8 waves. The 16xK A tile is staged once into LDS
// (async) with row pitch K+8 halves (bank stride 4 -> conflict-minimal
// ds_load_b128). Each wave computes a 16x64 tile (4 f32 accumulators, A
// fragment reused 4x). grid.x = ceil(N/512), grid.y = M/16.
// ---------------------------------------------------------------------------
__global__ __launch_bounds__(256)
void gemm_wmma(const _Float16* __restrict__ In, int ldin,
               const _Float16* __restrict__ W, int ldw,
               const float* __restrict__ bias,
               float* __restrict__ Out, int ldo,
               int N, int K,
               const int* __restrict__ dl, int t)
{
    extern __shared__ _Float16 sA[];          // 16 * (K + 8) halves
    const int lane = threadIdx.x & 31;
    const int wave = threadIdx.x >> 5;
    const int hi   = lane >> 4;               // 0 or 1 (half-wave)
    const int lr   = lane & 15;
    const int m0   = blockIdx.y << 4;
    const int n0   = blockIdx.x * 512 + wave * 64;
    const int pitch = K + 8;

    // --- stage A tile (16 rows x K halves) into LDS --------------------
    const int cpr   = K >> 3;                 // 16B chunks per row
    const int total = 16 * cpr;               // multiple of 256 for K%128==0
    for (int idx = threadIdx.x; idx < total; idx += 256) {
        int row = idx / cpr;
        int ch  = idx - row * cpr;
        stage_b128(In + (size_t)(m0 + row) * ldin + ch * 8,
                   sA + (size_t)row * pitch + ch * 8);
    }
    stage_wait();
    __syncthreads();

    // --- B row pointers (row-clamped gather; OOB cols never stored) ----
    const _Float16* brow[4];
    #pragma unroll
    for (int q = 0; q < 4; ++q) {
        int n = n0 + q * 16 + lr;
        if (n > N - 1) n = N - 1;
        brow[q] = W + (size_t)n * ldw;
    }

    v8f acc[4] = {};
    const _Float16* arow = sA + (size_t)lr * pitch;

    for (int k = 0; k < K; k += 32) {
        // A fragment from LDS: element j -> K = k + (j<8 ? j : j+8) + 8*hi
        const _Float16* pa = arow + k + 8 * hi;
        v8h a0 = *(const v8h*)pa;             // ds_load_b128
        v8h a1 = *(const v8h*)(pa + 16);
        v16h af;
        #pragma unroll
        for (int i = 0; i < 8; ++i) { af[i] = a0[i]; af[i + 8] = a1[i]; }

        if (k + 64 < K) __builtin_prefetch(brow[0] + k + 64, 0, 3);

        // B fragments: element j -> K = k + (j<8 ? j : j+8) + 8*hi
        #pragma unroll
        for (int q = 0; q < 4; ++q) {
            const _Float16* pb = brow[q] + k + 8 * hi;
            v8h b0 = *(const v8h*)pb;
            v8h b1 = *(const v8h*)(pb + 16);
            v16h bf;
            #pragma unroll
            for (int i = 0; i < 8; ++i) { bf[i] = b0[i]; bf[i + 8] = b1[i]; }
            acc[q] = __builtin_amdgcn_wmma_f32_16x16x32_f16(
                         false, af, false, bf, (short)0, acc[q], false, false);
        }
    }

    // --- epilogue: C/D layout M = r + 8*(lane/16), N = lane%16 ---------
    #pragma unroll
    for (int q = 0; q < 4; ++q) {
        const int col = n0 + q * 16 + lr;
        if (col >= N) continue;
        const float bv = bias ? bias[col] : 0.0f;
        #pragma unroll
        for (int r = 0; r < 8; ++r) {
            int row = m0 + r + 8 * hi;
            float mask = 1.0f;
            if (dl) mask = (t < dl[row]) ? 1.0f : 0.0f;
            Out[(size_t)row * ldo + col] = (acc[q][r] + bv) * mask;
        }
    }
}

static inline void launch_gemm(hipStream_t s,
                               const _Float16* In, int ldin,
                               const _Float16* W, int ldw,
                               const float* bias,
                               float* Out, int ldo,
                               int M, int N, int K,
                               const int* dl, int t)
{
    dim3 grid((N + 511) / 512, M / 16);
    size_t shmem = (size_t)16 * (K + 8) * sizeof(_Float16);
    gemm_wmma<<<grid, dim3(256), shmem, s>>>(In, ldin, W, ldw, bias, Out, ldo,
                                             N, K, dl, t);
}

// ---------------------------------------------------------------------------
// Small helper kernels
// ---------------------------------------------------------------------------
__global__ void cvt_f16(const float* __restrict__ in, _Float16* __restrict__ out, int n)
{
    int i = blockIdx.x * 256 + threadIdx.x;
    if (i < n) out[i] = (_Float16)in[i];
}

__global__ void build_wcat(const float* __restrict__ w_ih,
                           const float* __restrict__ w_hh,
                           _Float16* __restrict__ w)
{
    int i = blockIdx.x * 256 + threadIdx.x;
    if (i >= kG * kKxh) return;
    int n = i / kKxh, k = i % kKxh;
    float v = (k < 2 * kE) ? w_ih[(size_t)n * (2 * kE) + k]
                           : w_hh[(size_t)n * kH + (k - 2 * kE)];
    w[i] = (_Float16)v;
}

__global__ void vec_add(const float* a, const float* b, float* o, int n)
{
    int i = blockIdx.x * 256 + threadIdx.x;
    if (i < n) o[i] = a[i] + b[i];
}

// Stable descending sort of caption lengths (B=128), gather captions,
// and emit the integer outputs (as float) for d_out.
__global__ void sort_and_gather(const int* __restrict__ cap_len,
                                const int* __restrict__ captions,
                                int* __restrict__ sort_ind,
                                int* __restrict__ dl,
                                int* __restrict__ cap_s,
                                float* __restrict__ out_caps,
                                float* __restrict__ out_dl,
                                float* __restrict__ out_si)
{
    __shared__ int scl[kB];
    int i = threadIdx.x;                 // blockDim.x == 128
    scl[i] = cap_len[i];
    __syncthreads();
    int cli = scl[i];
    int rank = 0;
    for (int j = 0; j < kB; ++j) {
        int c = scl[j];
        if (c > cli || (c == cli && j < i)) rank++;
    }
    sort_ind[rank] = i;
    __syncthreads();
    int src = sort_ind[i];
    int d = scl[src] - 1;
    dl[i]      = d;
    out_dl[i]  = (float)d;
    out_si[i]  = (float)src;
    for (int l = 0; l < kL; ++l) {
        int tok = captions[src * kL + l];
        cap_s[i * kL + l]    = tok;
        out_caps[i * kL + l] = (float)tok;
    }
}

__global__ void gather_spatial(const float* __restrict__ sp,
                               const int* __restrict__ si,
                               float* __restrict__ out32,
                               _Float16* __restrict__ out16)
{
    int i = blockIdx.x * 256 + threadIdx.x;
    if (i >= kB * kP * kH) return;
    int b = i / (kP * kH);
    int rest = i % (kP * kH);
    float v = sp[(size_t)si[b] * kP * kH + rest];
    out32[i] = v;
    out16[i] = (_Float16)v;
}

__global__ void gather_global(const float* __restrict__ gf,
                              const int* __restrict__ si,
                              float* __restrict__ out)
{
    int i = blockIdx.x * 256 + threadIdx.x;
    if (i >= kB * kE) return;
    int b = i / kE;
    out[i] = gf[(size_t)si[b] * kE + (i % kE)];
}

__global__ void mean_enc_k(const float* __restrict__ enc,
                           const int* __restrict__ si,
                           _Float16* __restrict__ out16)
{
    int i = blockIdx.x * 256 + threadIdx.x;
    if (i >= kB * kIF) return;
    int b = i / kIF, f = i % kIF;
    const float* base = enc + (size_t)si[b] * kP * kIF + f;
    float s = 0.0f;
    for (int p = 0; p < kP; ++p) s += base[(size_t)p * kIF];
    out16[i] = (_Float16)(s * (1.0f / (float)kP));
}

__global__ void build_xh(const float* __restrict__ emb_W,
                         const int* __restrict__ cap_s, int t,
                         const float* __restrict__ glob,
                         const float* __restrict__ h,
                         _Float16* __restrict__ xh)
{
    int i = blockIdx.x * 256 + threadIdx.x;
    if (i >= kB * kKxh) return;
    int b = i / kKxh, k = i % kKxh;
    float v;
    if (k < kE)           v = emb_W[(size_t)cap_s[b * kL + t] * kE + k];
    else if (k < 2 * kE)  v = glob[(size_t)b * kE + (k - kE)];
    else                  v = h[(size_t)b * kH + (k - 2 * kE)];
    xh[i] = (_Float16)v;
}

__device__ __forceinline__ float sigf(float x) { return 1.0f / (1.0f + expf(-x)); }

__global__ void lstm_cell(const float* __restrict__ gates,
                          const float* __restrict__ c,
                          float* __restrict__ c_new,
                          float* __restrict__ h_new,
                          float* __restrict__ s_new,
                          _Float16* __restrict__ hn16,
                          _Float16* __restrict__ sn16)
{
    int i = blockIdx.x * 256 + threadIdx.x;
    if (i >= kB * kH) return;
    int b = i / kH, j = i % kH;
    const float* g = gates + (size_t)b * kG;
    float ig = sigf(g[j]);
    float fg = sigf(g[kH + j]);
    float gg = tanhf(g[2 * kH + j]);
    float og = sigf(g[3 * kH + j]);
    float sg = sigf(g[4 * kH + j]);
    float cn = fg * c[i] + ig * gg;
    float tc = tanhf(cn);
    float hn = og * tc;
    float sn = sg * tc;
    c_new[i] = cn;
    h_new[i] = hn;
    s_new[i] = sn;
    hn16[i] = (_Float16)hn;
    sn16[i] = (_Float16)sn;
}

// One block per (sorted) batch row: attention scores, softmaxes, context,
// adaptive gate beta, fc input, masked h/c state update, alpha/beta outputs.
__global__ __launch_bounds__(256)
void attention_step(const float* __restrict__ dec,    // [B,64] (49 valid)
                    const float* __restrict__ sent,   // [B,64]
                    const float* __restrict__ encatt, // [B*P,64] (49 valid)
                    const float* __restrict__ v,      // [49]
                    const float* __restrict__ spatial,// [B,P,H] f32 sorted
                    const float* __restrict__ h_new,
                    const float* __restrict__ s_new,
                    const float* __restrict__ c_new,
                    float* __restrict__ h,
                    float* __restrict__ c,
                    const int* __restrict__ dl, int t,
                    float* __restrict__ alphas_out,   // [B,T,P]
                    float* __restrict__ betas_out,    // [B,T]
                    _Float16* __restrict__ fcin16)    // [B,H]
{
    __shared__ float sdec[kA], ssent[kA], szt[kP + 1], salpha[kP];
    __shared__ float sbeta_sh;
    const int tid = threadIdx.x;
    const int b   = blockIdx.x;

    if (tid < kA) {
        sdec[tid]  = dec[b * 64 + tid];
        ssent[tid] = sent[b * 64 + tid];
    }
    __syncthreads();

    if (tid < kP) {
        const float* ea = encatt + ((size_t)b * kP + tid) * 64;
        float z = 0.0f;
        for (int a = 0; a < kA; ++a) z += tanhf(ea[a] + sdec[a]) * v[a];
        szt[tid] = z;
    }
    if (tid == kP) {
        float z = 0.0f;
        for (int a = 0; a < kA; ++a) z += tanhf(ssent[a] + sdec[a]) * v[a];
        szt[kP] = z;
    }
    __syncthreads();

    if (tid == 0) {
        float mx = szt[0];
        for (int p = 1; p < kP; ++p) mx = fmaxf(mx, szt[p]);
        float s = 0.0f;
        for (int p = 0; p < kP; ++p) { float e = expf(szt[p] - mx); salpha[p] = e; s += e; }
        float inv = 1.0f / s;
        for (int p = 0; p < kP; ++p) salpha[p] *= inv;
        // alpha_hat over [zt, temp] (P+1 elements); beta = last component
        float mx2 = fmaxf(mx, szt[kP]);
        float s2 = 0.0f;
        for (int p = 0; p <= kP; ++p) s2 += expf(szt[p] - mx2);
        sbeta_sh = expf(szt[kP] - mx2) / s2;
    }
    __syncthreads();

    const float m = (t < dl[b]) ? 1.0f : 0.0f;
    const float beta = sbeta_sh;
    if (tid < kP) alphas_out[((size_t)b * kT + t) * kP + tid] = salpha[tid] * m;
    if (tid == 0) betas_out[(size_t)b * kT + t] = beta * m;

    for (int j = tid; j < kH; j += blockDim.x) {
        size_t idx = (size_t)b * kH + j;
        float ct = 0.0f;
        const float* spb = spatial + (size_t)b * kP * kH + j;
        for (int p = 0; p < kP; ++p) ct += salpha[p] * spb[(size_t)p * kH];
        float chat = beta * s_new[idx] + (1.0f - beta) * ct;
        fcin16[idx] = (_Float16)(chat + h_new[idx]);
        h[idx] = m * h_new[idx] + (1.0f - m) * h[idx];
        c[idx] = m * c_new[idx] + (1.0f - m) * c[idx];
    }
}

// ---------------------------------------------------------------------------
// Host orchestration
// ---------------------------------------------------------------------------
extern "C" void kernel_launch(void* const* d_in, const int* in_sizes, int n_in,
                              void* d_out, int out_size, void* d_ws, size_t ws_size,
                              hipStream_t stream)
{
    (void)in_sizes; (void)n_in; (void)out_size; (void)ws_size;

    const float* spatial_f  = (const float*)d_in[0];
    const float* global_f   = (const float*)d_in[1];
    const float* enc_image  = (const float*)d_in[2];
    const int*   captions   = (const int*)d_in[3];
    const int*   cap_len    = (const int*)d_in[4];
    const float* emb_W      = (const float*)d_in[5];
    const float* w_ih       = (const float*)d_in[6];
    const float* w_hh       = (const float*)d_in[7];
    const float* b_ih       = (const float*)d_in[8];
    const float* b_hh       = (const float*)d_in[9];
    const float* fc_W       = (const float*)d_in[10];
    const float* fc_b       = (const float*)d_in[11];
    const float* enc_att_W  = (const float*)d_in[12];
    const float* enc_att_b  = (const float*)d_in[13];
    const float* dec_att_W  = (const float*)d_in[14];
    const float* sent_att_W = (const float*)d_in[15];
    const float* att_v      = (const float*)d_in[16];
    const float* init_h_W   = (const float*)d_in[17];
    const float* init_h_b   = (const float*)d_in[18];
    const float* init_c_W   = (const float*)d_in[19];
    const float* init_c_b   = (const float*)d_in[20];

    // Output layout: preds | alphas | betas | captions | decode_lengths | sort_ind
    float* out       = (float*)d_out;
    float* out_preds = out;
    float* out_alph  = out_preds + (size_t)kB * kT * kV;
    float* out_beta  = out_alph  + (size_t)kB * kT * kP;
    float* out_caps  = out_beta  + (size_t)kB * kT;
    float* out_dl    = out_caps  + (size_t)kB * kL;
    float* out_si    = out_dl    + kB;

    // Workspace bump allocator (d_ws is 256B-aligned from hipMalloc)
    char* wptr = (char*)d_ws;
    auto alloc = [&](size_t bytes) -> void* {
        void* p = (void*)wptr;
        wptr += (bytes + 255) & ~(size_t)255;
        return p;
    };
    int*      i_si    = (int*)alloc((size_t)kB * 4);
    int*      i_dl    = (int*)alloc((size_t)kB * 4);
    int*      i_caps  = (int*)alloc((size_t)kB * kL * 4);
    float*    f_glob  = (float*)alloc((size_t)kB * kE * 4);
    float*    f_spat  = (float*)alloc((size_t)kB * kP * kH * 4);
    _Float16* h_spat  = (_Float16*)alloc((size_t)kB * kP * kH * 2);
    _Float16* h_mean  = (_Float16*)alloc((size_t)kB * kIF * 2);
    float*    f_encatt= (float*)alloc((size_t)kB * kP * 64 * 4);
    float*    f_h     = (float*)alloc((size_t)kB * kH * 4);
    float*    f_c     = (float*)alloc((size_t)kB * kH * 4);
    float*    f_cn    = (float*)alloc((size_t)kB * kH * 4);
    float*    f_hn    = (float*)alloc((size_t)kB * kH * 4);
    float*    f_sn    = (float*)alloc((size_t)kB * kH * 4);
    float*    f_gates = (float*)alloc((size_t)kB * kG * 4);
    float*    f_dec   = (float*)alloc((size_t)kB * 64 * 4);
    float*    f_sent  = (float*)alloc((size_t)kB * 64 * 4);
    float*    f_bsum  = (float*)alloc((size_t)kG * 4);
    _Float16* h_wcat  = (_Float16*)alloc((size_t)kG * kKxh * 2);
    _Float16* h_fc    = (_Float16*)alloc((size_t)kV * kH * 2);
    _Float16* h_ihw   = (_Float16*)alloc((size_t)kH * kIF * 2);
    _Float16* h_icw   = (_Float16*)alloc((size_t)kH * kIF * 2);
    _Float16* h_eaw   = (_Float16*)alloc((size_t)kA * kH * 2);
    _Float16* h_daw   = (_Float16*)alloc((size_t)kA * kH * 2);
    _Float16* h_saw   = (_Float16*)alloc((size_t)kA * kH * 2);
    _Float16* h_xh    = (_Float16*)alloc((size_t)kB * kKxh * 2);
    _Float16* h_hn16  = (_Float16*)alloc((size_t)kB * kH * 2);
    _Float16* h_sn16  = (_Float16*)alloc((size_t)kB * kH * 2);
    _Float16* h_fcin  = (_Float16*)alloc((size_t)kB * kH * 2);

    auto cgrid = [](long n) { return dim3((unsigned)((n + 255) / 256)); };

    // Weight conversion (f32 -> f16) and bias precompute
    cvt_f16<<<cgrid((long)kV * kH), 256, 0, stream>>>(fc_W, h_fc, kV * kH);
    cvt_f16<<<cgrid((long)kH * kIF), 256, 0, stream>>>(init_h_W, h_ihw, kH * kIF);
    cvt_f16<<<cgrid((long)kH * kIF), 256, 0, stream>>>(init_c_W, h_icw, kH * kIF);
    cvt_f16<<<cgrid((long)kA * kH), 256, 0, stream>>>(enc_att_W, h_eaw, kA * kH);
    cvt_f16<<<cgrid((long)kA * kH), 256, 0, stream>>>(dec_att_W, h_daw, kA * kH);
    cvt_f16<<<cgrid((long)kA * kH), 256, 0, stream>>>(sent_att_W, h_saw, kA * kH);
    build_wcat<<<cgrid((long)kG * kKxh), 256, 0, stream>>>(w_ih, w_hh, h_wcat);
    vec_add<<<cgrid(kG), 256, 0, stream>>>(b_ih, b_hh, f_bsum, kG);

    // Sort by caption length (descending, stable) + gathers
    sort_and_gather<<<1, kB, 0, stream>>>(cap_len, captions, i_si, i_dl, i_caps,
                                          out_caps, out_dl, out_si);
    gather_spatial<<<cgrid((long)kB * kP * kH), 256, 0, stream>>>(spatial_f, i_si, f_spat, h_spat);
    gather_global<<<cgrid((long)kB * kE), 256, 0, stream>>>(global_f, i_si, f_glob);
    mean_enc_k<<<cgrid((long)kB * kIF), 256, 0, stream>>>(enc_image, i_si, h_mean);

    // h0 = mean_enc @ init_h_W^T + b ; c0 likewise   (WMMA GEMMs)
    launch_gemm(stream, h_mean, kIF, h_ihw, kIF, init_h_b, f_h, kH, kB, kH, kIF, nullptr, 0);
    launch_gemm(stream, h_mean, kIF, h_icw, kIF, init_c_b, f_c, kH, kB, kH, kIF, nullptr, 0);

    // enc_att_out = spatial_sorted @ enc_att_W^T + b   (M = B*P = 6272)
    launch_gemm(stream, h_spat, kH, h_eaw, kH, enc_att_b, f_encatt, 64,
                kB * kP, kA, kH, nullptr, 0);

    // Sequential decode
    for (int t = 0; t < kT; ++t) {
        build_xh<<<cgrid((long)kB * kKxh), 256, 0, stream>>>(emb_W, i_caps, t, f_glob, f_h, h_xh);
        // gates = [x,h] @ [w_ih|w_hh]^T + (b_ih + b_hh)
        launch_gemm(stream, h_xh, kKxh, h_wcat, kKxh, f_bsum, f_gates, kG,
                    kB, kG, kKxh, nullptr, 0);
        lstm_cell<<<cgrid((long)kB * kH), 256, 0, stream>>>(f_gates, f_c, f_cn, f_hn, f_sn,
                                                            h_hn16, h_sn16);
        launch_gemm(stream, h_hn16, kH, h_daw, kH, nullptr, f_dec, 64, kB, kA, kH, nullptr, 0);
        launch_gemm(stream, h_sn16, kH, h_saw, kH, nullptr, f_sent, 64, kB, kA, kH, nullptr, 0);
        attention_step<<<kB, 256, 0, stream>>>(f_dec, f_sent, f_encatt, att_v, f_spat,
                                               f_hn, f_sn, f_cn, f_h, f_c, i_dl, t,
                                               out_alph, out_beta, h_fcin);
        // preds[:, t, :] = (c_hat + h_new) @ fc_W^T + fc_b, masked
        launch_gemm(stream, h_fcin, kH, h_fc, kH, fc_b, out_preds + (size_t)t * kV,
                    kT * kV, kB, kV, kH, i_dl, t);
    }
}
